// NeuralCDE_84361747628489
// MI455X (gfx1250) — matrix operand order repaired and checked by reading, not compiled, not run on hardware
//
#include <hip/hip_runtime.h>
#include <hip/hip_bf16.h>

typedef unsigned short u16;
typedef __attribute__((ext_vector_type(16))) __bf16 v16bf;
typedef __attribute__((ext_vector_type(8)))  __bf16 v8bf;
typedef __attribute__((ext_vector_type(8)))  float  v8f;

// ---------- scalar helpers ----------
__device__ __forceinline__ u16 f2bf(float f) {
    unsigned u = __builtin_bit_cast(unsigned, f);
    unsigned r = (u + 0x7FFFu + ((u >> 16) & 1u)) >> 16;
    return (u16)r;
}
__device__ __forceinline__ float bf2f(u16 h) {
    unsigned u = ((unsigned)h) << 16;
    return __builtin_bit_cast(float, u);
}
__device__ __forceinline__ float fast_tanh(float x) {
#if __has_builtin(__builtin_amdgcn_tanhf)
    return __builtin_amdgcn_tanhf(x);
#elif __has_builtin(__builtin_amdgcn_tanh_f32)
    return __builtin_amdgcn_tanh_f32(x);
#else
    return tanhf(x);
#endif
}

// A fragment (16x32 bf16, MxK) from row-major bf16 LDS buffer with leading dim `lda`.
// 16-bit A layout: lane l holds row m=l%16; K groups {8*(l/16)..+8} and {16+8*(l/16)..+8}.
__device__ __forceinline__ v16bf load_a(const u16* buf, int lda, int mbase, int kbase, int lane) {
    const int m = mbase + (lane & 15);
    const int g = (lane >> 4) * 8;
    const v8bf lo = *(const v8bf*)(buf + m * lda + kbase + g);
    const v8bf hi = *(const v8bf*)(buf + m * lda + kbase + g + 16);
    v16bf a;
#pragma unroll
    for (int i = 0; i < 8; ++i) { a[i] = lo[i]; a[i + 8] = hi[i]; }
    return a;
}
// B fragment (32x16 bf16, KxN) pre-swizzled in memory: lane l holds column n=l%16, its 16 K-values contiguously in table k order.
__device__ __forceinline__ v16bf load_b(const u16* frag, int lane) {
    return *(const v16bf*)(frag + lane * 16);
}
__device__ __forceinline__ v8f wmma_bf16(v16bf a, v16bf b, v8f c) {
    return __builtin_amdgcn_wmma_f32_16x16x32_bf16(false, a, false, b, (short)0, c, false, false);
}

// ---------- weight repack kernels (fragment order, bf16) ----------
__global__ void pack_plain_kernel(const float* __restrict__ src, u16* __restrict__ dst, int K, int N) {
    int tid = blockIdx.x * 256 + threadIdx.x;
    if (tid >= K * N) return;
    int f = tid >> 9, r = tid & 511;
    int ln = r >> 4, s = r & 15, nl = ln & 15;
    int kl = (s & 7) + ((ln >> 4) << 3) + ((s >> 3) << 4);
    int ktiles = K >> 5;
    int nt = f / ktiles, kt = f % ktiles;
    dst[tid] = f2bf(src[(kt * 32 + kl) * N + nt * 16 + nl]);
}
// W3 (128 x 4224, col index = h*33+c) -> W3r frags [(nt*33 + c)*4 + kt][lane][s] (n=lane%16, k in table order), K=(c,w), N=h
__global__ void pack_w3_kernel(const float* __restrict__ src, u16* __restrict__ dst) {
    int tid = blockIdx.x * 256 + threadIdx.x;
    if (tid >= 4224 * 128) return;
    int f = tid >> 9, r = tid & 511;
    int ln = r >> 4, s = r & 15, nl = ln & 15;
    int kl = (s & 7) + ((ln >> 4) << 3) + ((s >> 3) << 4);
    int kt = f & 3, c = (f >> 2) % 33, nt = f / 132;
    int w = kt * 32 + kl, h = nt * 16 + nl;
    dst[tid] = f2bf(src[(size_t)w * 4224 + h * 33 + c]);
}

// ---------- persistent CDE scan ----------
#define BT   32
#define LDA  136   // 128 + 8 pad (keeps 16B align, breaks bank conflicts)

// one 128x128 bf16 GEMM layer: dst = tanh(src @ Wf + bias); wave handles n-tile = wave.
__device__ __forceinline__ void gemm_tanh_layer(const u16* src, const u16* __restrict__ wf,
                                                const float* __restrict__ bias, u16* dst,
                                                int wave, int lane) {
    const int nloc = lane & 15, half8 = (lane >> 4) * 8;
    const float bv = bias[wave * 16 + nloc];
#pragma unroll
    for (int t = 0; t < 2; ++t) {
        v8f acc = {};
#pragma unroll
        for (int kt = 0; kt < 4; ++kt) {
            v16bf A = load_a(src, LDA, t * 16, kt * 32, lane);
            v16bf B = load_b(wf + ((size_t)wave * 4 + kt) * 512, lane);
            acc = wmma_bf16(A, B, acc);
        }
#pragma unroll
        for (int r = 0; r < 8; ++r) {
            int m = t * 16 + r + half8;
            dst[m * LDA + wave * 16 + nloc] = f2bf(fast_tanh(acc[r] + bv));
        }
    }
}

__global__ __launch_bounds__(256) void cde_scan_kernel(
    const float* __restrict__ coeffs,                               // (512,127,132)
    const float* __restrict__ iW1, const float* __restrict__ ib1,   // 33x128
    const float* __restrict__ iW2, const float* __restrict__ ib2,   // 128x128
    const float* __restrict__ iW3, const float* __restrict__ ib3,   // 128x128
    const u16* __restrict__ w1f, const float* __restrict__ fb1,
    const u16* __restrict__ w2f, const float* __restrict__ fb2,
    const u16* __restrict__ w3f, const float* __restrict__ fb3,     // fb3: 4224
    u16* __restrict__ zseq)                                         // (512*128, 128) bf16
{
    __shared__ u16   bufA[BT * LDA];
    __shared__ u16   bufB[BT * LDA];
    __shared__ float dxl[BT * 34];
    __shared__ float pbl[BT * 34], pcl[BT * 34], pdl[BT * 34];
    __shared__ float b3r[33 * 128];

    const int tid = threadIdx.x;
    const int wave = tid >> 5, lane = tid & 31;
    const int nloc = lane & 15, half8 = (lane >> 4) * 8;
    const int ncol = wave * 16 + nloc;
    const int bb = blockIdx.x * BT;

    // repacked func-layer-3 bias: b3r[c][h] = fb3[h*33+c]
    for (int i = tid; i < 33 * 128; i += 256) { int c = i >> 7, h = i & 127; b3r[i] = fb3[h * 33 + c]; }

    // ---- init MLP on a0 = coeffs[:,0,0:33]  (33->128 tanh, 128->128 tanh, 128->128 tanh) ----
    for (int i = tid; i < BT * 128; i += 256) {
        int m = i >> 7, n = i & 127;
        const float* arow = coeffs + (size_t)(bb + m) * 127 * 132;
        float s = ib1[n];
        for (int k = 0; k < 33; ++k) s += arow[k] * iW1[k * 128 + n];
        bufB[m * LDA + n] = f2bf(fast_tanh(s));
    }
    __syncthreads();
    for (int i = tid; i < BT * 128; i += 256) {
        int m = i >> 7, n = i & 127;
        float s = ib2[n];
        for (int k = 0; k < 128; ++k) s += bf2f(bufB[m * LDA + k]) * iW2[k * 128 + n];
        bufA[m * LDA + n] = f2bf(fast_tanh(s));
    }
    __syncthreads();
    for (int i = tid; i < BT * 128; i += 256) {
        int m = i >> 7, n = i & 127;
        float s = ib3[n];
        for (int k = 0; k < 128; ++k) s += bf2f(bufA[m * LDA + k]) * iW3[k * 128 + n];
        bufB[m * LDA + n] = f2bf(fast_tanh(s));
    }
    __syncthreads();

    // per-lane state: z, k1..k3 (wave owns h-columns [wave*16, +16), both 16-row M tiles)
    float z[16], k1[16], k2[16], k3[16];
#pragma unroll
    for (int t = 0; t < 2; ++t)
#pragma unroll
        for (int r = 0; r < 8; ++r) {
            int m = t * 16 + r + half8;
            float v = bf2f(bufB[m * LDA + ncol]);
            z[t * 8 + r] = v;
            zseq[((size_t)(bb + m) * 128 + 0) * 128 + ncol] = f2bf(v);
        }
    __syncthreads();

    for (int p = 0; p < 127; ++p) {
        // spline coeffs of this piece: b, 2c, 3d
        for (int i = tid; i < BT * 33; i += 256) {
            int m = i / 33, c = i % 33;
            const float* base = coeffs + ((size_t)(bb + m) * 127 + p) * 132;
            pbl[m * 34 + c] = base[33 + c];
            pcl[m * 34 + c] = base[66 + c];
            pdl[m * 34 + c] = base[99 + c];
        }
        __syncthreads();

        float g[16];
        for (int e = 0; e < 4; ++e) {
            const float frac = (float)e * (1.0f / 3.0f);
            // z input of this RK4 stage (3/8 rule), staged to LDS as bf16
#pragma unroll
            for (int t = 0; t < 2; ++t)
#pragma unroll
                for (int r = 0; r < 8; ++r) {
                    int i = t * 8 + r;
                    float zi = z[i];
                    if (e == 1) zi += k1[i] * (1.0f / 3.0f);
                    else if (e == 2) zi += k2[i] - k1[i] * (1.0f / 3.0f);
                    else if (e == 3) zi += k1[i] - k2[i] + k3[i];
                    int m = t * 16 + r + half8;
                    bufA[m * LDA + ncol] = f2bf(zi);
                }
            // dX(frac) = b + (2c + 3d*frac)*frac
            for (int i = tid; i < BT * 33; i += 256) {
                int m = i / 33, c = i % 33;
                int o = m * 34 + c;
                dxl[o] = pbl[o] + (pcl[o] + pdl[o] * frac) * frac;
            }
            __syncthreads();

            gemm_tanh_layer(bufA, w1f, fb1, bufB, wave, lane);   // h1
            __syncthreads();
            gemm_tanh_layer(bufB, w2f, fb2, bufA, wave, lane);   // h2
            __syncthreads();

            // ---- GEMM3: g[b,h] = sum_c dX[b,c] * ( (h2 @ W3r[c])[b,h] + b3r[c,h] ) ----
            v16bf Af[8];
#pragma unroll
            for (int t = 0; t < 2; ++t)
#pragma unroll
                for (int kt = 0; kt < 4; ++kt)
                    Af[t * 4 + kt] = load_a(bufA, LDA, t * 16, kt * 32, lane);

#pragma unroll
            for (int i = 0; i < 16; ++i) g[i] = 0.0f;

            const u16* wp = w3f + (size_t)wave * 33 * 4 * 512;   // this wave's h-tile column
            for (int c = 0; c < 33; ++c) {
                const u16* wc = wp + (size_t)c * 2048;
                if (c < 32) __builtin_prefetch((const void*)(wc + 2048 + lane * 16), 0, 3);
                v8f a0 = {}, a1 = {};
#pragma unroll
                for (int kt = 0; kt < 4; ++kt) {
                    v16bf B = load_b(wc + kt * 512, lane);
                    a0 = wmma_bf16(Af[kt],     B, a0);
                    a1 = wmma_bf16(Af[4 + kt], B, a1);
                }
                const float bc = b3r[c * 128 + ncol];
#pragma unroll
                for (int r = 0; r < 8; ++r) {
                    g[r]     += dxl[(r + half8) * 34 + c]      * (a0[r] + bc);
                    g[8 + r] += dxl[(16 + r + half8) * 34 + c] * (a1[r] + bc);
                }
            }

            if (e == 0) { for (int i = 0; i < 16; ++i) k1[i] = g[i]; }
            else if (e == 1) { for (int i = 0; i < 16; ++i) k2[i] = g[i]; }
            else if (e == 2) { for (int i = 0; i < 16; ++i) k3[i] = g[i]; }
            else {
#pragma unroll
                for (int i = 0; i < 16; ++i)
                    z[i] += (k1[i] + 3.0f * (k2[i] + k3[i]) + g[i]) * 0.125f;
            }
            __syncthreads();
        }
        // emit z_{p+1}
#pragma unroll
        for (int t = 0; t < 2; ++t)
#pragma unroll
            for (int r = 0; r < 8; ++r) {
                int m = t * 16 + r + half8;
                zseq[((size_t)(bb + m) * 128 + (p + 1)) * 128 + ncol] = f2bf(z[t * 8 + r]);
            }
    }
}

// ---------- seq2seq decoder: (65536 x 128) -> tanh(128->64) -> (64->64) ----------
#define DLDA 136
#define HLDA 72

__global__ __launch_bounds__(256) void cde_decode_kernel(
    const u16* __restrict__ zseq,
    const u16* __restrict__ d1f, const float* __restrict__ db1,
    const u16* __restrict__ d2f, const float* __restrict__ db2,
    float* __restrict__ out)
{
    __shared__ u16 xb[128 * DLDA];
    __shared__ u16 hb[128 * HLDA];
    const int tid = threadIdx.x;
    const int wave = tid >> 5, lane = tid & 31;
    const int nloc = lane & 15, half8 = (lane >> 4) * 8;
    const size_t rowbase = (size_t)blockIdx.x * 128;

    for (int i = tid; i < 128 * 16; i += 256) {
        int row = i >> 4, ch = i & 15;
        *(uint4*)(xb + row * DLDA + ch * 8) = *(const uint4*)(zseq + (rowbase + row) * 128 + ch * 8);
    }
    __syncthreads();

#pragma unroll
    for (int nt = 0; nt < 4; ++nt) {
        v8f acc = {};
#pragma unroll
        for (int kt = 0; kt < 4; ++kt) {
            v16bf A = load_a(xb, DLDA, wave * 16, kt * 32, lane);
            v16bf B = load_b(d1f + ((size_t)nt * 4 + kt) * 512, lane);
            acc = wmma_bf16(A, B, acc);
        }
        float bv = db1[nt * 16 + nloc];
#pragma unroll
        for (int r = 0; r < 8; ++r)
            hb[(wave * 16 + r + half8) * HLDA + nt * 16 + nloc] = f2bf(fast_tanh(acc[r] + bv));
    }
    __syncthreads();

#pragma unroll
    for (int nt = 0; nt < 4; ++nt) {
        v8f acc = {};
#pragma unroll
        for (int kt = 0; kt < 2; ++kt) {
            v16bf A = load_a(hb, HLDA, wave * 16, kt * 32, lane);
            v16bf B = load_b(d2f + ((size_t)nt * 2 + kt) * 512, lane);
            acc = wmma_bf16(A, B, acc);
        }
        float bv = db2[nt * 16 + nloc];
#pragma unroll
        for (int r = 0; r < 8; ++r)
            out[(rowbase + wave * 16 + r + half8) * 64 + nt * 16 + nloc] = acc[r] + bv;
    }
}

// ---------- host ----------
extern "C" void kernel_launch(void* const* d_in, const int* in_sizes, int n_in,
                              void* d_out, int out_size, void* d_ws, size_t ws_size,
                              hipStream_t stream) {
    const float* coeffs = (const float*)d_in[0];
    const float* iW1 = (const float*)d_in[1];  const float* ib1 = (const float*)d_in[2];
    const float* iW2 = (const float*)d_in[3];  const float* ib2 = (const float*)d_in[4];
    const float* iW3 = (const float*)d_in[5];  const float* ib3 = (const float*)d_in[6];
    const float* fW1 = (const float*)d_in[7];  const float* fb1 = (const float*)d_in[8];
    const float* fW2 = (const float*)d_in[9];  const float* fb2 = (const float*)d_in[10];
    const float* fW3 = (const float*)d_in[11]; const float* fb3 = (const float*)d_in[12];
    const float* dW1 = (const float*)d_in[13]; const float* db1 = (const float*)d_in[14];
    const float* dW2 = (const float*)d_in[15]; const float* db2 = (const float*)d_in[16];

    size_t off = 0;
    auto alloc = [&](size_t bytes) { size_t r = off; off += (bytes + 255) & ~(size_t)255; return r; };
    u16* w1f  = (u16*)((char*)d_ws + alloc(16384 * 2));
    u16* w2f  = (u16*)((char*)d_ws + alloc(16384 * 2));
    u16* w3f  = (u16*)((char*)d_ws + alloc((size_t)540672 * 2));
    u16* d1f  = (u16*)((char*)d_ws + alloc(8192 * 2));
    u16* d2f  = (u16*)((char*)d_ws + alloc(4096 * 2));
    u16* zseq = (u16*)((char*)d_ws + alloc((size_t)512 * 128 * 128 * 2));
    (void)ws_size; (void)n_in; (void)in_sizes; (void)out_size;

    pack_plain_kernel<<<16384 / 256, 256, 0, stream>>>(fW1, w1f, 128, 128);
    pack_plain_kernel<<<16384 / 256, 256, 0, stream>>>(fW2, w2f, 128, 128);
    pack_w3_kernel  <<<540672 / 256, 256, 0, stream>>>(fW3, w3f);
    pack_plain_kernel<<<8192 / 256, 256, 0, stream>>>(dW1, d1f, 128, 64);
    pack_plain_kernel<<<4096 / 256, 256, 0, stream>>>(dW2, d2f, 64, 64);

    cde_scan_kernel<<<16, 256, 0, stream>>>(coeffs, iW1, ib1, iW2, ib2, iW3, ib3,
                                            w1f, fb1, w2f, fb2, w3f, fb3, zseq);
    cde_decode_kernel<<<512, 256, 0, stream>>>(zseq, d1f, db1, d2f, db2, (float*)d_out);
}
